// GPT_3358664426112
// MI455X (gfx1250) — matrix operand order repaired and checked by reading, not compiled, not run on hardware
//
#include <hip/hip_runtime.h>
#include <hip/hip_bf16.h>

// ---------------- constants (GPT-2 small, B=4, S=1024) ----------------
#define GL  12        // layers
#define GH  12        // heads
#define GD  768       // model dim
#define GHD 64        // head dim
#define GV  50257     // vocab
#define GB  4
#define GS  1024
#define GM  (GB*GS)   // 4096 token rows
#define QKV_N (3*GD)  // 2304
#define KOFF GD       // 768
#define VOFF (2*GD)   // 1536

typedef __attribute__((ext_vector_type(16))) _Float16 v16h;
typedef __attribute__((ext_vector_type(8)))  float    v8f;
typedef __attribute__((ext_vector_type(4)))  unsigned int u32x4;
typedef __attribute__((ext_vector_type(4)))  float    f32x4;
typedef __attribute__((ext_vector_type(4)))  unsigned int ui32x4;
typedef __attribute__((ext_vector_type(8)))  int      si32x8;
typedef __attribute__((ext_vector_type(4)))  int      si32x4;

union HFrag {
    v16h v;
    unsigned int u[8];
    u32x4 q[2];
};

__device__ inline unsigned lds_addr_of(const void* p) {
    return (unsigned)(unsigned long long)(size_t)p;
}

// Issue a TDM DMA: 2D tile of f16, tile_dim0=32 elems/row (64B),
// tile_dim1=128 rows, row stride K elems, LDS pitch 80B (64B + 16B pad).
__device__ inline void tdm_load_a_tile(const _Float16* gptr, unsigned ldsOff,
                                       int K, int M)
{
    unsigned long long ga = (unsigned long long)(size_t)gptr;
    ui32x4 g0;
    g0[0] = 1u;                                          // count=1, user desc
    g0[1] = ldsOff;                                      // LDS byte address
    g0[2] = (unsigned)(ga & 0xFFFFFFFFu);                // global addr lo
    g0[3] = (unsigned)((ga >> 32) & 0x01FFFFFFu)         // global addr hi
          | 0x80000000u;                                 // type=2 ("image")
    unsigned flags = (1u << 16)    // data_size = 2 bytes
                   | (1u << 20)    // pad_enable
                   | (3u << 22)    // pad_interval: 16 dwords (64B)
                   | (3u << 25);   // pad_amount: 4 dwords (16B)
    si32x8 g1;
    g1[0] = (int)flags;
    g1[1] = (int)(((unsigned)K & 0xFFFFu) << 16);                 // dim0 lo16
    g1[2] = (int)(((unsigned)K >> 16) | (((unsigned)M & 0xFFFFu) << 16));
    g1[3] = (int)(((unsigned)M >> 16) | (32u << 16));             // tile0=32
    g1[4] = (int)128u;                                            // tile1=128
    g1[5] = (int)(unsigned)K;                                     // stride0
    g1[6] = 0;
    g1[7] = 0;
    si32x4 z4 = {};
    si32x8 z8 = {};
    __builtin_amdgcn_tensor_load_to_lds(g0, g1, z4, z4, z8, 0);
}

// ======================================================================
// Embedding: x[b,s,:] = wte[idx[b,s],:] + wpe[s,:]   (fp32 residual)
// ======================================================================
__global__ __launch_bounds__(256)
void embed_kernel(const int* __restrict__ idx, const float* __restrict__ wte,
                  const float* __restrict__ wpe, float* __restrict__ x, int n)
{
    int i = blockIdx.x * blockDim.x + threadIdx.x;
    if (i >= n) return;
    int col = i % GD;
    int row = i / GD;
    int pos = row % GS;
    x[i] = wte[(size_t)idx[row] * GD + col] + wpe[(size_t)pos * GD + col];
}

// ======================================================================
// LayerNorm (fp32 in, f16 out)
// ======================================================================
__global__ __launch_bounds__(256)
void layernorm_f16(const float* __restrict__ x, const float* __restrict__ w,
                   const float* __restrict__ b, _Float16* __restrict__ out)
{
    __shared__ float sred[256];
    int row = blockIdx.x;
    int tid = threadIdx.x;
    const float* xr = x + (size_t)row * GD;

    float v0 = xr[tid], v1 = xr[tid + 256], v2 = xr[tid + 512];
    sred[tid] = v0 + v1 + v2;
    __syncthreads();
    for (int st = 128; st > 0; st >>= 1) {
        if (tid < st) sred[tid] += sred[tid + st];
        __syncthreads();
    }
    float mu = sred[0] * (1.0f / GD);
    __syncthreads();

    float d0 = v0 - mu, d1 = v1 - mu, d2 = v2 - mu;
    sred[tid] = d0 * d0 + d1 * d1 + d2 * d2;
    __syncthreads();
    for (int st = 128; st > 0; st >>= 1) {
        if (tid < st) sred[tid] += sred[tid + st];
        __syncthreads();
    }
    float rstd = rsqrtf(sred[0] * (1.0f / GD) + 1e-5f);

    _Float16* orow = out + (size_t)row * GD;
    orow[tid]       = (_Float16)(d0 * rstd * w[tid]       + b[tid]);
    orow[tid + 256] = (_Float16)(d1 * rstd * w[tid + 256] + b[tid + 256]);
    orow[tid + 512] = (_Float16)(d2 * rstd * w[tid + 512] + b[tid + 512]);
}

// ======================================================================
// WMMA GEMM:  out = act( A[M,K](f16) * W[K,N](fp32->f16) + bias ) + res
// 128x128 block tile, BK=32, 8 waves (2x4), wave tile 64x32.
// A tile staged by TDM DMA (double buffered, overlapped with compute).
// W tile software-pipelined through VGPRs with on-the-fly f32->f16.
// ======================================================================
#define BM 128
#define BN 128
#define BK 32
#define APITCH 40  // halves: 64B row + 16B pad (matches TDM descriptor)
#define BPITCH 40  // halves per LDS B row (B stored n-major, k-minor)

__global__ __launch_bounds__(256)
void gemm_wmma(const _Float16* __restrict__ A, const float* __restrict__ W,
               const float* __restrict__ bias, const float* __restrict__ res,
               float* __restrict__ outF, _Float16* __restrict__ outH,
               int M, int N, int K, int doGelu)
{
    __shared__ _Float16 As[2][BM * APITCH];   // [row][k]
    __shared__ _Float16 Bs[2][BN * BPITCH];   // [n][k] (k-minor for B-frags)

    int tid  = threadIdx.x;
    int lane = tid & 31;
    int wave = tid >> 5;
    int wm   = wave >> 2;   // 0..1  (64 rows each)
    int wn   = wave & 3;    // 0..3  (32 cols each)
    int g    = lane >> 4;
    int ln   = lane & 15;

    int blockM = blockIdx.y * BM;
    int blockN = blockIdx.x * BN;
    const _Float16* Abase = A + (size_t)blockM * K;
    bool nAligned = ((N & 3) == 0);

    v8f zero = {};
    v8f acc[4][2];
    for (int i = 0; i < 4; ++i)
        for (int j = 0; j < 2; ++j) acc[i][j] = zero;

    // W-tile staging registers (software pipeline)
    float wreg[16];
    int   wr[16], wc[16];
    for (int i = 0; i < 16; ++i) {
        int eid = tid + i * 256;
        wr[i] = eid >> 7;      // k row 0..31
        wc[i] = eid & 127;     // n col 0..127
    }

    // ---- prologue: tile 0 ----
    if (wave == 0)
        tdm_load_a_tile(Abase, lds_addr_of(&As[0][0]), K, M);
    if (nAligned) {
        for (int i = 0; i < 4; ++i) {
            int fid = tid + i * 256;
            int r   = fid >> 5;
            int c4  = (fid & 31) * 4;
            f32x4 v = *(const f32x4*)&W[(size_t)r * N + blockN + c4];
            Bs[0][(c4 + 0) * BPITCH + r] = (_Float16)v[0];
            Bs[0][(c4 + 1) * BPITCH + r] = (_Float16)v[1];
            Bs[0][(c4 + 2) * BPITCH + r] = (_Float16)v[2];
            Bs[0][(c4 + 3) * BPITCH + r] = (_Float16)v[3];
        }
    } else {
        for (int i = 0; i < 16; ++i) {
            int n = blockN + wc[i];
            float wv = (n < N) ? W[(size_t)wr[i] * N + n] : 0.0f;
            Bs[0][wc[i] * BPITCH + wr[i]] = (_Float16)wv;
        }
    }
    if (wave == 0) __builtin_amdgcn_s_wait_tensorcnt(0);
    __syncthreads();

    int nsteps = K / BK;
    int cur = 0;
    for (int it = 0; it < nsteps; ++it) {
        int  nxt     = cur ^ 1;
        bool hasNext = (it + 1 < nsteps);
        int  kn      = (it + 1) * BK;

        // ---- early-issue next tile loads (overlap with WMMA) ----
        f32x4 wv4[4];
        if (hasNext) {
            if (wave == 0)
                tdm_load_a_tile(Abase + kn, lds_addr_of(&As[nxt][0]), K, M);
            if (nAligned) {
                for (int i = 0; i < 4; ++i) {
                    int fid = tid + i * 256;
                    int r   = fid >> 5;
                    int c4  = (fid & 31) * 4;
                    wv4[i]  = *(const f32x4*)&W[(size_t)(kn + r) * N + blockN + c4];
                }
            } else {
                for (int i = 0; i < 16; ++i) {
                    int n = blockN + wc[i];
                    wreg[i] = (n < N) ? W[(size_t)(kn + wr[i]) * N + n] : 0.0f;
                }
            }
        }

        // ---- compute on current tile ----
        HFrag bf[2];
        for (int nt = 0; nt < 2; ++nt) {
            const _Float16* bp = &Bs[cur][(wn * 32 + nt * 16 + ln) * BPITCH + g * 8];
            bf[nt].q[0] = *(const u32x4*)bp;
            bf[nt].q[1] = *(const u32x4*)(bp + 16);
        }
        for (int mt = 0; mt < 4; ++mt) {
            HFrag af;
            const _Float16* ap = &As[cur][(wm * 64 + mt * 16 + ln) * APITCH + g * 8];
            af.q[0] = *(const u32x4*)ap;
            af.q[1] = *(const u32x4*)(ap + 16);
            for (int nt = 0; nt < 2; ++nt)
                acc[mt][nt] = __builtin_amdgcn_wmma_f32_16x16x32_f16(
                    false, af.v, false, bf[nt].v, (short)0, acc[mt][nt],
                    false, false);
        }

        // ---- retire next-tile W into LDS, sync for buffer swap ----
        if (hasNext) {
            if (nAligned) {
                for (int i = 0; i < 4; ++i) {
                    int fid = tid + i * 256;
                    int r   = fid >> 5;
                    int c4  = (fid & 31) * 4;
                    Bs[nxt][(c4 + 0) * BPITCH + r] = (_Float16)wv4[i][0];
                    Bs[nxt][(c4 + 1) * BPITCH + r] = (_Float16)wv4[i][1];
                    Bs[nxt][(c4 + 2) * BPITCH + r] = (_Float16)wv4[i][2];
                    Bs[nxt][(c4 + 3) * BPITCH + r] = (_Float16)wv4[i][3];
                }
            } else {
                for (int i = 0; i < 16; ++i)
                    Bs[nxt][wc[i] * BPITCH + wr[i]] = (_Float16)wreg[i];
            }
            if (wave == 0) __builtin_amdgcn_s_wait_tensorcnt(0);
            __syncthreads();
        }
        cur = nxt;
    }

    // ---- epilogue: bias, GELU, residual, stores ----
    for (int mt = 0; mt < 4; ++mt) {
        for (int nt = 0; nt < 2; ++nt) {
            int n = blockN + wn * 32 + nt * 16 + ln;
            if (n >= N) continue;
            float bv = bias ? bias[n] : 0.0f;
            for (int r = 0; r < 8; ++r) {
                int m = blockM + wm * 64 + mt * 16 + g * 8 + r;
                float v = acc[mt][nt][r] + bv;
                if (doGelu) v = 0.5f * v * (1.0f + erff(v * 0.70710678118f));
                if (res)  v += res[(size_t)m * N + n];
                if (outF) outF[(size_t)m * N + n] = v;
                if (outH) outH[(size_t)m * N + n] = (_Float16)v;
            }
        }
    }
}

// ======================================================================
// Flash attention (causal), one wave per (b, h, 16-query tile).
// Q*K^T and P*V via v_wmma_f32_16x16x32_f16, online softmax.
// ======================================================================
#define PPITCH 40
#define VPITCH 40

__global__ __launch_bounds__(32)
void attention_wmma(const _Float16* __restrict__ qkv, _Float16* __restrict__ o)
{
    __shared__ _Float16 pLDS[16 * PPITCH];   // P tile [m][k]
    __shared__ _Float16 vLDS[64 * VPITCH];   // V chunk transposed [d][key]

    int lane = threadIdx.x;
    int g  = lane >> 4;
    int ln = lane & 15;
    int qt = blockIdx.x;
    int h  = blockIdx.y;
    int b  = blockIdx.z;
    int q0 = qt * 16;
    size_t base = (size_t)b * GS * QKV_N;

    // --- Q A-frags: two 16x32 chunks, each = two b128 loads
    HFrag aQ[2];
    {
        const _Float16* qp = qkv + base + (size_t)(q0 + ln) * QKV_N + h * GHD;
        for (int c = 0; c < 2; ++c) {
            aQ[c].q[0] = *(const u32x4*)(qp + c * 32 + g * 8);
            aQ[c].q[1] = *(const u32x4*)(qp + c * 32 + 16 + g * 8);
        }
    }

    float rmax[8], rsum[8];
    v8f zero = {};
    v8f O[4];
    for (int r = 0; r < 8; ++r) { rmax[r] = -1e30f; rsum[r] = 0.0f; }
    for (int nt = 0; nt < 4; ++nt) O[nt] = zero;

    int cmax = (q0 + 15) >> 5;
    for (int c = 0; c <= cmax; ++c) {
        int k0 = c * 32;

        // --- S = Q*K^T for two 16-key tiles
        v8f sAcc[2];
        for (int t = 0; t < 2; ++t) {
            HFrag bK[2];
            const _Float16* kp = qkv + base +
                (size_t)(k0 + t * 16 + ln) * QKV_N + KOFF + h * GHD;
            for (int dc = 0; dc < 2; ++dc) {
                bK[dc].q[0] = *(const u32x4*)(kp + dc * 32 + g * 8);
                bK[dc].q[1] = *(const u32x4*)(kp + dc * 32 + 16 + g * 8);
            }
            v8f s = zero;
            s = __builtin_amdgcn_wmma_f32_16x16x32_f16(
                    false, aQ[0].v, false, bK[0].v, (short)0, s, false, false);
            s = __builtin_amdgcn_wmma_f32_16x16x32_f16(
                    false, aQ[1].v, false, bK[1].v, (short)0, s, false, false);
            sAcc[t] = s;
        }

        // --- scale + causal mask
        float s0[8], s1[8];
        for (int r = 0; r < 8; ++r) {
            int m = g * 8 + r;
            s0[r] = (k0 + ln      <= q0 + m) ? sAcc[0][r] * 0.125f : -1e30f;
            s1[r] = (k0 + 16 + ln <= q0 + m) ? sAcc[1][r] * 0.125f : -1e30f;
        }

        // --- stage V chunk transposed into LDS (b128 loads)
        for (int i = 0; i < 8; ++i) {
            int chunk = lane + i * 32;      // 256 chunks of 8 halves
            int key   = chunk >> 3;
            int dp    = (chunk & 7) * 8;
            union { u32x4 q; _Float16 hh[8]; } cv;
            cv.q = *(const u32x4*)
                (qkv + base + (size_t)(k0 + key) * QKV_N + VOFF + h * GHD + dp);
            for (int j = 0; j < 8; ++j)
                vLDS[(dp + j) * VPITCH + key] = cv.hh[j];
        }

        // --- online softmax update
        float corr[8];
        for (int r = 0; r < 8; ++r) {
            float mx = fmaxf(s0[r], s1[r]);
            for (int off = 8; off >= 1; off >>= 1)
                mx = fmaxf(mx, __shfl_xor(mx, off, 32));
            float newm = fmaxf(rmax[r], mx);
            float p0 = expf(s0[r] - newm);
            float p1 = expf(s1[r] - newm);
            float ps = p0 + p1;
            for (int off = 8; off >= 1; off >>= 1)
                ps += __shfl_xor(ps, off, 32);
            float co = expf(rmax[r] - newm);
            rsum[r] = rsum[r] * co + ps;
            rmax[r] = newm;
            corr[r] = co;
            pLDS[(g * 8 + r) * PPITCH + ln]      = (_Float16)p0;
            pLDS[(g * 8 + r) * PPITCH + 16 + ln] = (_Float16)p1;
        }
        __syncthreads();

        // --- P as A-frag, V as B-frags; O = P*V + O (rescaled)
        HFrag aP;
        {
            const _Float16* pp = &pLDS[ln * PPITCH + g * 8];
            aP.q[0] = *(const u32x4*)pp;
            aP.q[1] = *(const u32x4*)(pp + 16);
        }
        for (int nt = 0; nt < 4; ++nt) {
            for (int r = 0; r < 8; ++r) O[nt][r] *= corr[r];
            HFrag bV;
            const _Float16* vp = &vLDS[(nt * 16 + ln) * VPITCH + g * 8];
            bV.q[0] = *(const u32x4*)vp;
            bV.q[1] = *(const u32x4*)(vp + 16);
            O[nt] = __builtin_amdgcn_wmma_f32_16x16x32_f16(
                        false, aP.v, false, bV.v, (short)0, O[nt], false, false);
        }
        __syncthreads();
    }

    // --- normalize + store o (f16)
    for (int nt = 0; nt < 4; ++nt)
        for (int r = 0; r < 8; ++r) {
            int m = g * 8 + r;
            o[(size_t)(b * GS + q0 + m) * GD + h * GHD + nt * 16 + ln] =
                (_Float16)(O[nt][r] / rsum[r]);
        }
}

// ======================================================================
// Cross-entropy: per-row log-sum-exp, then deterministic mean reduce.
// ======================================================================
__global__ __launch_bounds__(256)
void row_loss_kernel(const float* __restrict__ logits,
                     const int* __restrict__ labels, float* __restrict__ rl)
{
    __shared__ float sred[256];
    int row = blockIdx.x;
    int tid = threadIdx.x;
    const float* z = logits + (size_t)row * GV;

    float mx = -3.0e38f;
    for (int j = tid; j < GV; j += 256) mx = fmaxf(mx, z[j]);
    sred[tid] = mx; __syncthreads();
    for (int st = 128; st > 0; st >>= 1) {
        if (tid < st) sred[tid] = fmaxf(sred[tid], sred[tid + st]);
        __syncthreads();
    }
    mx = sred[0]; __syncthreads();

    float sm = 0.0f;
    for (int j = tid; j < GV; j += 256) sm += expf(z[j] - mx);
    sred[tid] = sm; __syncthreads();
    for (int st = 128; st > 0; st >>= 1) {
        if (tid < st) sred[tid] += sred[tid + st];
        __syncthreads();
    }
    if (tid == 0) {
        float lse = logf(sred[0]) + mx;
        rl[row] = lse - z[labels[row]];
    }
}

__global__ __launch_bounds__(256)
void loss_reduce_kernel(const float* __restrict__ rl, float* __restrict__ out)
{
    __shared__ float sred[256];
    int tid = threadIdx.x;
    float s = 0.0f;
    for (int j = tid; j < GM; j += 256) s += rl[j];
    sred[tid] = s; __syncthreads();
    for (int st = 128; st > 0; st >>= 1) {
        if (tid < st) sred[tid] += sred[tid + st];
        __syncthreads();
    }
    if (tid == 0) out[0] = sred[0] * (1.0f / GM);
}

// ======================================================================
// Host orchestration
// ======================================================================
extern "C" void kernel_launch(void* const* d_in, const int* in_sizes, int n_in,
                              void* d_out, int out_size, void* d_ws, size_t ws_size,
                              hipStream_t stream)
{
    (void)in_sizes; (void)n_in; (void)out_size; (void)ws_size;

    const int*   idx    = (const int*)  d_in[0];
    const int*   pred   = (const int*)  d_in[1];
    const float* wte    = (const float*)d_in[2];
    const float* wpe    = (const float*)d_in[3];
    const float* ln1_w  = (const float*)d_in[4];
    const float* ln1_b  = (const float*)d_in[5];
    const float* qkv_w  = (const float*)d_in[6];
    const float* qkv_b  = (const float*)d_in[7];
    const float* proj_w = (const float*)d_in[8];
    const float* proj_b = (const float*)d_in[9];
    const float* ln2_w  = (const float*)d_in[10];
    const float* ln2_b  = (const float*)d_in[11];
    const float* f1_w   = (const float*)d_in[12];
    const float* f1_b   = (const float*)d_in[13];
    const float* f2_w   = (const float*)d_in[14];
    const float* f2_b   = (const float*)d_in[15];
    const float* lnf_w  = (const float*)d_in[16];
    const float* lnf_b  = (const float*)d_in[17];
    const float* lm_w   = (const float*)d_in[18];

    float* logits = (float*)d_out;                       // [GM, GV]
    float* lossp  = logits + (size_t)GM * GV;            // final scalar

    char* ws = (char*)d_ws;
    size_t off = 0;
    auto takeup = [&](size_t bytes) {
        char* p = ws + off;
        off = (off + bytes + 255) & ~(size_t)255;
        return p;
    };
    float*     x     = (float*)    takeup((size_t)GM * GD * 4);      // residual
    _Float16*  hbuf  = (_Float16*) takeup((size_t)GM * GD * 2);      // LN out
    _Float16*  qkvb  = (_Float16*) takeup((size_t)GM * QKV_N * 2);   // qkv f16
    _Float16*  obuf  = (_Float16*) takeup((size_t)GM * GD * 2);      // attn out
    _Float16*  m1buf = (_Float16*) takeup((size_t)GM * 4 * GD * 2);  // MLP mid
    float*     rl    = (float*)    takeup((size_t)GM * 4);           // row loss

    // 1) embedding
    embed_kernel<<<(GM * GD + 255) / 256, 256, 0, stream>>>(idx, wte, wpe, x, GM * GD);

    // 2) transformer blocks
    for (int l = 0; l < GL; ++l) {
        layernorm_f16<<<GM, 256, 0, stream>>>(x, ln1_w + (size_t)l * GD,
                                              ln1_b + (size_t)l * GD, hbuf);
        gemm_wmma<<<dim3(QKV_N / BN, GM / BM), 256, 0, stream>>>(
            hbuf, qkv_w + (size_t)l * GD * QKV_N, qkv_b + (size_t)l * QKV_N,
            nullptr, nullptr, qkvb, GM, QKV_N, GD, 0);

        attention_wmma<<<dim3(GS / 16, GH, GB), 32, 0, stream>>>(qkvb, obuf);

        gemm_wmma<<<dim3(GD / BN, GM / BM), 256, 0, stream>>>(
            obuf, proj_w + (size_t)l * GD * GD, proj_b + (size_t)l * GD,
            x, x, nullptr, GM, GD, GD, 0);

        layernorm_f16<<<GM, 256, 0, stream>>>(x, ln2_w + (size_t)l * GD,
                                              ln2_b + (size_t)l * GD, hbuf);
        gemm_wmma<<<dim3(4 * GD / BN, GM / BM), 256, 0, stream>>>(
            hbuf, f1_w + (size_t)l * GD * 4 * GD, f1_b + (size_t)l * 4 * GD,
            nullptr, nullptr, m1buf, GM, 4 * GD, GD, 1);
        gemm_wmma<<<dim3(GD / BN, GM / BM), 256, 0, stream>>>(
            m1buf, f2_w + (size_t)l * 4 * GD * GD, f2_b + (size_t)l * GD,
            x, x, nullptr, GM, GD, 4 * GD, 0);
    }

    // 3) final LN + LM head
    layernorm_f16<<<GM, 256, 0, stream>>>(x, lnf_w, lnf_b, hbuf);
    gemm_wmma<<<dim3((GV + BN - 1) / BN, GM / BM), 256, 0, stream>>>(
        hbuf, lm_w, nullptr, nullptr, logits, nullptr, GM, GV, GD, 0);

    // 4) cross-entropy loss
    row_loss_kernel<<<GM, 256, 0, stream>>>(logits, pred, rl);
    loss_reduce_kernel<<<1, 256, 0, stream>>>(rl, lossp);
}
